// SpatialSelfAttention_48292612276562
// MI455X (gfx1250) — compile-verified, matrix operand order checked by reading, not run on hardware
//
#include <hip/hip_runtime.h>
#include <hip/hip_bf16.h>

// ---------------------------------------------------------------------------
// Spatial self-attention for MI455X (gfx1250), wave32, WMMA bf16 pipeline.
// B=16, C=512, H=W=32 (HW=1024).
// Round 2: transposed-score softmax (no cross-lane transpose) + async
// global->LDS double-buffered K staging (CDNA5 async path).
// ---------------------------------------------------------------------------

#define BATCH 16
#define CCH   512
#define HWPX  1024
#define KSTEPS (CCH / 32)   // 16 k-steps of 32 per 512-deep dot product
#define SCALE  0.044194173824159216f  // 512^-0.5

typedef __bf16 bf16;
typedef __attribute__((ext_vector_type(16))) __bf16 v16bf;
typedef __attribute__((ext_vector_type(8)))  __bf16 v8bf;
typedef __attribute__((ext_vector_type(8)))  float  v8f;

static __device__ __forceinline__ v8f zero8() {
    v8f z;
#pragma unroll
    for (int i = 0; i < 8; ++i) z[i] = 0.0f;
    return z;
}

static __device__ __forceinline__ v8f wmma_bf16(v16bf a, v16bf b, v8f c) {
    return __builtin_amdgcn_wmma_f32_16x16x32_bf16(
        false, a, false, b, (short)0, c, false, false);
}

// 16x32 bf16 fragment load (A layout; also B 32x16 when storage is
// K-contiguous per column). Per ISA 7.12.2:
//   lane m (<16): K runs {0..7, 16..23};  lane m+16: K runs {8..15, 24..31}
static __device__ __forceinline__ v16bf load_frag(const bf16* base, int ld) {
    const int lane = threadIdx.x & 31;
    const int row  = lane & 15;
    const int sel  = (lane >> 4) << 3;  // 0 or 8
    const bf16* p  = base + (size_t)row * ld + sel;
    v8bf lo = *(const v8bf*)(p);
    v8bf hi = *(const v8bf*)(p + 16);
    v16bf r;
#pragma unroll
    for (int i = 0; i < 8; ++i) { r[i] = lo[i]; r[i + 8] = hi[i]; }
    return r;
}

// Async-copy one 16-row K tile (16KB, contiguous) into LDS.
// 32 x global_load_async_to_lds_b128: each moves 32 lanes x 16B = 512B.
// IOFFSET applies to both the LDS and global addresses (ISA 10.  async), so a
// single base VGPR pair + immediate offsets covers the whole tile.
static __device__ __forceinline__ void stage_k_async(const bf16* src, const bf16* lds_dst) {
    const int lane = threadIdx.x & 31;
    const unsigned int laddr = (unsigned int)(size_t)(lds_dst) + lane * 16u;
    const char* g = (const char*)src + lane * 16;
#pragma unroll
    for (int i = 0; i < 32; ++i) {
        asm volatile("global_load_async_to_lds_b128 %0, %1, off offset:%2"
                     :: "v"(laddr), "v"(g), "i"(i * 512) : "memory");
    }
}

static __device__ __forceinline__ void wait_async_le32() {
    asm volatile("s_wait_asynccnt 0x20" ::: "memory");
}
static __device__ __forceinline__ void wait_async_0() {
    asm volatile("s_wait_asynccnt 0x0" ::: "memory");
}
static __device__ __forceinline__ void wait_ds_0() {
    asm volatile("s_wait_dscnt 0x0" ::: "memory");
}

// ---------------------------------------------------------------------------
// prep: x [B][C][HW] f32 -> xT [B][HW][C] bf16 ; W [O][C] f32 -> bf16 (x4)
// ---------------------------------------------------------------------------
__global__ void prep_x_kernel(const float* __restrict__ x, bf16* __restrict__ xT) {
    size_t idx = (size_t)blockIdx.x * blockDim.x + threadIdx.x;  // (b,p,c), c fastest
    if (idx >= (size_t)BATCH * HWPX * CCH) return;
    int c = (int)(idx % CCH);
    int p = (int)((idx / CCH) % HWPX);
    int b = (int)(idx / ((size_t)CCH * HWPX));
    xT[idx] = (bf16)x[(size_t)b * CCH * HWPX + (size_t)c * HWPX + p];
}

__global__ void prep_w_kernel(const float* __restrict__ wq, const float* __restrict__ wk,
                              const float* __restrict__ wv, const float* __restrict__ wo,
                              bf16* __restrict__ wqb, bf16* __restrict__ wkb,
                              bf16* __restrict__ wvb, bf16* __restrict__ wob) {
    size_t idx = (size_t)blockIdx.x * blockDim.x + threadIdx.x;
    if (idx >= (size_t)CCH * CCH) return;
    wqb[idx] = (bf16)wq[idx];
    wkb[idx] = (bf16)wk[idx];
    wvb[idx] = (bf16)wv[idx];
    wob[idx] = (bf16)wo[idx];
}

// ---------------------------------------------------------------------------
// Q/K projection, pixel-major out: Y[p][o] = sum_c xT[p][c] W[o][c] + bias[o]
// grid: (C/16, HW/16, B), block 32
// ---------------------------------------------------------------------------
__global__ void __launch_bounds__(32)
qk_proj_kernel(const bf16* __restrict__ xT,
               const bf16* __restrict__ Wqb, const float* __restrict__ bq,
               const bf16* __restrict__ Wkb, const float* __restrict__ bk,
               bf16* __restrict__ Q, bf16* __restrict__ K) {
    const int ot = blockIdx.x, pt = blockIdx.y, b = blockIdx.z;
    const bf16* xb = xT + (size_t)b * HWPX * CCH + (size_t)pt * 16 * CCH;
    v8f accq = zero8(), acck = zero8();
#pragma unroll 4
    for (int kk = 0; kk < KSTEPS; ++kk) {
        v16bf a  = load_frag(xb + kk * 32, CCH);
        v16bf fq = load_frag(Wqb + (size_t)ot * 16 * CCH + kk * 32, CCH);
        v16bf fk = load_frag(Wkb + (size_t)ot * 16 * CCH + kk * 32, CCH);
        accq = wmma_bf16(a, fq, accq);
        acck = wmma_bf16(a, fk, acck);
    }
    const int lane = threadIdx.x & 31;
    const int n = lane & 15, mb = (lane >> 4) << 3;
    const float bqv = bq[ot * 16 + n];
    const float bkv = bk[ot * 16 + n];
#pragma unroll
    for (int r = 0; r < 8; ++r) {
        size_t off = (size_t)b * HWPX * CCH + (size_t)(pt * 16 + mb + r) * CCH + ot * 16 + n;
        Q[off] = (bf16)(accq[r] + bqv);
        K[off] = (bf16)(acck[r] + bkv);
    }
}

// ---------------------------------------------------------------------------
// V projection, channel-major out: Vt[o][p] = sum_c W[o][c] xT[p][c] + bias[o]
// grid: (HW/16, C/16, B)
// ---------------------------------------------------------------------------
__global__ void __launch_bounds__(32)
v_proj_kernel(const bf16* __restrict__ xT,
              const bf16* __restrict__ Wvb, const float* __restrict__ bv,
              bf16* __restrict__ Vt) {
    const int pt = blockIdx.x, ot = blockIdx.y, b = blockIdx.z;
    v8f acc = zero8();
#pragma unroll 4
    for (int kk = 0; kk < KSTEPS; ++kk) {
        v16bf a  = load_frag(Wvb + (size_t)ot * 16 * CCH + kk * 32, CCH);
        v16bf bx = load_frag(xT + (size_t)b * HWPX * CCH + (size_t)pt * 16 * CCH + kk * 32, CCH);
        acc = wmma_bf16(a, bx, acc);
    }
    const int lane = threadIdx.x & 31;
    const int n = lane & 15, mb = (lane >> 4) << 3;
#pragma unroll
    for (int r = 0; r < 8; ++r) {
        int o = ot * 16 + mb + r;
        Vt[(size_t)b * CCH * HWPX + (size_t)o * HWPX + pt * 16 + n] = (bf16)(acc[r] + bv[o]);
    }
}

// ---------------------------------------------------------------------------
// Attention, one wave per (batch, 16-query tile).
// Computes S^T = K * Q^T so each lane owns 8 entries of ONE softmax row
// (i = lane&15); exp'd probabilities are natively in A-fragment layout.
// K tiles double-buffered in LDS via async global->LDS DMA.
// grid: (B, HW/16)
// ---------------------------------------------------------------------------
__global__ void __launch_bounds__(32)
attn_kernel(const bf16* __restrict__ Q, const bf16* __restrict__ K,
            const bf16* __restrict__ Vt, bf16* __restrict__ AOt) {
    const int b = blockIdx.x;
    const int itile = blockIdx.y;
    const int lane = threadIdx.x & 31;
    const int n = lane & 15, mb = (lane >> 4) << 3;

    const bf16* Qb = Q  + (size_t)b * HWPX * CCH + (size_t)itile * 16 * CCH;
    const bf16* Kb = K  + (size_t)b * HWPX * CCH;
    const bf16* Vb = Vt + (size_t)b * CCH * HWPX;

    __shared__ __align__(16) bf16 Klds[2][16 * CCH];   // 2 x 16KB double buffer
    __shared__ __align__(16) bf16 Plds[64 * 32 * 8];   // 32KB, per-lane frag layout

    // Preload Q B-fragments: B[kk=c][n=i] = Q[i][c], K-contiguous per column.
    v16bf qf[KSTEPS];
#pragma unroll
    for (int kk = 0; kk < KSTEPS; ++kk) qf[kk] = load_frag(Qb + kk * 32, CCH);

    // ---- pass 1: row max (rows are i = lane&15; lanes L / L+16 share a row)
    float rmax = -3.0e38f;
    stage_k_async(Kb, &Klds[0][0]);
    for (int jt = 0; jt < 63; ++jt) {
        stage_k_async(Kb + (size_t)(jt + 1) * 16 * CCH, &Klds[(jt + 1) & 1][0]);
        wait_async_le32();                       // current tile resident
        const bf16* Kt = &Klds[jt & 1][0];
        v8f s = zero8();
#pragma unroll 4
        for (int kk = 0; kk < KSTEPS; ++kk)
            s = wmma_bf16(load_frag(Kt + kk * 32, CCH), qf[kk], s);
#pragma unroll
        for (int r = 0; r < 8; ++r) rmax = fmaxf(rmax, s[r] * SCALE);
        wait_ds_0();                             // frags consumed before reuse
    }
    {
        wait_async_0();
        const bf16* Kt = &Klds[63 & 1][0];
        v8f s = zero8();
#pragma unroll 4
        for (int kk = 0; kk < KSTEPS; ++kk)
            s = wmma_bf16(load_frag(Kt + kk * 32, CCH), qf[kk], s);
#pragma unroll
        for (int r = 0; r < 8; ++r) rmax = fmaxf(rmax, s[r] * SCALE);
        wait_ds_0();
    }
    rmax = fmaxf(rmax, __shfl_xor(rmax, 16, 32));

    // ---- pass 2: P = exp(S - max) -> LDS (A-fragment layout), row sums
    float rsum = 0.0f;
    stage_k_async(Kb, &Klds[0][0]);
    for (int jt = 0; jt < 63; ++jt) {
        stage_k_async(Kb + (size_t)(jt + 1) * 16 * CCH, &Klds[(jt + 1) & 1][0]);
        wait_async_le32();
        const bf16* Kt = &Klds[jt & 1][0];
        v8f s = zero8();
#pragma unroll 4
        for (int kk = 0; kk < KSTEPS; ++kk)
            s = wmma_bf16(load_frag(Kt + kk * 32, CCH), qf[kk], s);
        v8bf pk;
#pragma unroll
        for (int r = 0; r < 8; ++r) {
            float p = __expf(s[r] * SCALE - rmax);
            rsum += p;
            pk[r] = (bf16)p;
        }
        *(v8bf*)(Plds + ((size_t)jt * 32 + lane) * 8) = pk;
        wait_ds_0();
    }
    {
        wait_async_0();
        const bf16* Kt = &Klds[63 & 1][0];
        v8f s = zero8();
#pragma unroll 4
        for (int kk = 0; kk < KSTEPS; ++kk)
            s = wmma_bf16(load_frag(Kt + kk * 32, CCH), qf[kk], s);
        v8bf pk;
#pragma unroll
        for (int r = 0; r < 8; ++r) {
            float p = __expf(s[r] * SCALE - rmax);
            rsum += p;
            pk[r] = (bf16)p;
        }
        *(v8bf*)(Plds + ((size_t)63 * 32 + lane) * 8) = pk;
    }
    rsum += __shfl_xor(rsum, 16, 32);
    const float rinv = 1.0f / rsum;
    // O-tile rows are m = mb+r; fetch their 1/sum from the owning lanes.
    float rinvm[8];
#pragma unroll
    for (int r = 0; r < 8; ++r) rinvm[r] = __shfl(rinv, mb + r, 32);

    __syncthreads();

    // ---- O = P * V over 4 channel groups of 128 ----
    for (int cg = 0; cg < 4; ++cg) {
        v8f acc[8];
#pragma unroll
        for (int t = 0; t < 8; ++t) acc[t] = zero8();
        for (int jc = 0; jc < HWPX / 32; ++jc) {
            // P A-fragment: lane-native 16B loads from the two 16-j subtiles
            v8bf lo = *(const v8bf*)(Plds + ((size_t)(2 * jc) * 32 + lane) * 8);
            v8bf hi = *(const v8bf*)(Plds + ((size_t)(2 * jc + 1) * 32 + lane) * 8);
            v16bf pa;
#pragma unroll
            for (int i = 0; i < 8; ++i) { pa[i] = lo[i]; pa[i + 8] = hi[i]; }
            const bf16* vbase = Vb + (size_t)(cg * 128) * HWPX + jc * 32;
#pragma unroll
            for (int t = 0; t < 8; ++t)
                acc[t] = wmma_bf16(pa, load_frag(vbase + (size_t)t * 16 * HWPX, HWPX), acc[t]);
        }
#pragma unroll
        for (int t = 0; t < 8; ++t) {
#pragma unroll
            for (int r = 0; r < 8; ++r) {
                size_t off = (size_t)b * HWPX * CCH
                           + (size_t)(itile * 16 + mb + r) * CCH + cg * 128 + t * 16 + n;
                AOt[off] = (bf16)(acc[t][r] * rinvm[r]);
            }
        }
    }
}

// ---------------------------------------------------------------------------
// Output projection + residual: out[b][o][p] = x + sum_c Wo[o][c] AO[c][p] + bo[o]
// grid: (HW/16, C/16, B)
// ---------------------------------------------------------------------------
__global__ void __launch_bounds__(32)
out_proj_kernel(const bf16* __restrict__ AOt,
                const bf16* __restrict__ Wob, const float* __restrict__ bo,
                const float* __restrict__ x, float* __restrict__ out) {
    const int pt = blockIdx.x, ot = blockIdx.y, b = blockIdx.z;
    v8f acc = zero8();
#pragma unroll 4
    for (int kk = 0; kk < KSTEPS; ++kk) {
        v16bf a  = load_frag(Wob + (size_t)ot * 16 * CCH + kk * 32, CCH);
        v16bf bb = load_frag(AOt + (size_t)b * HWPX * CCH + (size_t)pt * 16 * CCH + kk * 32, CCH);
        acc = wmma_bf16(a, bb, acc);
    }
    const int lane = threadIdx.x & 31;
    const int n = lane & 15, mb = (lane >> 4) << 3;
#pragma unroll
    for (int r = 0; r < 8; ++r) {
        int o = ot * 16 + mb + r;
        size_t off = (size_t)b * CCH * HWPX + (size_t)o * HWPX + pt * 16 + n;
        out[off] = x[off] + acc[r] + bo[o];
    }
}

// ---------------------------------------------------------------------------
extern "C" void kernel_launch(void* const* d_in, const int* in_sizes, int n_in,
                              void* d_out, int out_size, void* d_ws, size_t ws_size,
                              hipStream_t stream) {
    const float* x  = (const float*)d_in[0];
    const float* Wq = (const float*)d_in[1];
    const float* bq = (const float*)d_in[2];
    const float* Wk = (const float*)d_in[3];
    const float* bk = (const float*)d_in[4];
    const float* Wv = (const float*)d_in[5];
    const float* bv = (const float*)d_in[6];
    const float* Wo = (const float*)d_in[7];
    const float* bo = (const float*)d_in[8];
    float* out = (float*)d_out;

    const size_t NX = (size_t)BATCH * HWPX * CCH;   // 8M bf16 = 16 MB
    const size_t NW = (size_t)CCH * CCH;            // 256K bf16 = 512 KB
    bf16* xT  = (bf16*)d_ws;
    bf16* Wqb = xT  + NX;
    bf16* Wkb = Wqb + NW;
    bf16* Wvb = Wkb + NW;
    bf16* Wob = Wvb + NW;
    bf16* Q   = Wob + NW;
    bf16* K   = Q   + NX;
    bf16* Vt  = K   + NX;
    bf16* AOt = Vt  + NX;

    {
        int threads = 256;
        int blocks = (int)((NX + threads - 1) / threads);
        prep_x_kernel<<<blocks, threads, 0, stream>>>(x, xT);
        blocks = (int)((NW + threads - 1) / threads);
        prep_w_kernel<<<blocks, threads, 0, stream>>>(Wq, Wk, Wv, Wo, Wqb, Wkb, Wvb, Wob);
    }

    qk_proj_kernel<<<dim3(CCH / 16, HWPX / 16, BATCH), 32, 0, stream>>>(
        xT, Wqb, bq, Wkb, bk, Q, K);

    v_proj_kernel<<<dim3(HWPX / 16, CCH / 16, BATCH), 32, 0, stream>>>(
        xT, Wvb, bv, Vt);

    attn_kernel<<<dim3(BATCH, HWPX / 16), 32, 0, stream>>>(Q, K, Vt, AOt);

    out_proj_kernel<<<dim3(HWPX / 16, CCH / 16, BATCH), 32, 0, stream>>>(
        AOt, Wob, bo, x, out);
}
